// AttentionBlock_25752623907152
// MI455X (gfx1250) — matrix operand order repaired and checked by reading, not compiled, not run on hardware
//
#include <hip/hip_runtime.h>
#include <hip/hip_bf16.h>

// ---------------------------------------------------------------------------
// MI455X (gfx1250) attention block:  out = Attn(x) @ Wo^T + bo + x
// v_wmma_f32_16x16x32_f16 for all matmuls + Tensor Data Mover LDS staging.
// ---------------------------------------------------------------------------

typedef __attribute__((ext_vector_type(16))) _Float16 v16h;
typedef __attribute__((ext_vector_type(8)))  _Float16 v8h;
typedef __attribute__((ext_vector_type(8)))  float    v8f;
typedef __attribute__((ext_vector_type(4)))  unsigned int u32x4;
typedef __attribute__((ext_vector_type(8)))  int      i32x8;
typedef __attribute__((ext_vector_type(4)))  int      i32x4;

union F16x16 { v16h v; v8h h[2]; };

#define WMMA_F16(A, B, C) \
    __builtin_amdgcn_wmma_f32_16x16x32_f16(false, (A), false, (B), (short)0, (C), false, false)

constexpr int Bsz = 4, Seq = 2048, Dm = 1024, Hn = 16, Dh = 64;
constexpr int Mrows = Bsz * Seq;            // 8192

// Workspace layout (in _Float16 elements)
constexpr unsigned long long XH_OFF   = 0ull;
constexpr unsigned long long WQKV_OFF = XH_OFF   + 8388608ull;
constexpr unsigned long long WO_OFF   = WQKV_OFF + 3145728ull;
constexpr unsigned long long QH_OFF   = WO_OFF   + 1048576ull;
constexpr unsigned long long KH_OFF   = QH_OFF   + 8388608ull;
constexpr unsigned long long VH_OFF   = KH_OFF   + 8388608ull;
constexpr unsigned long long AH_OFF   = VH_OFF   + 8388608ull;

// ---------------------------------------------------------------------------
// TDM: 2D tile load global->LDS.  data_size=2B.  tensor dims == tile dims
// (global_addr is the tile start, so no OOB).  pad codes per ISA 8.4:
// pad_interval: dwords-before-pad = 2^(code+1); pad_amount: dwords-1.
// ---------------------------------------------------------------------------
__device__ __forceinline__ unsigned int lds_lo32(const void* p) {
    return (unsigned int)(unsigned long long)p;
}

__device__ __forceinline__ void tdm_load_2d(unsigned int lds_byte, const void* gptr,
                                            unsigned int tile_d0, unsigned int tile_d1,
                                            unsigned long long stride0,
                                            unsigned int pad_interval_code,
                                            unsigned int pad_amount_code) {
    unsigned long long ga = (unsigned long long)gptr;
    u32x4 g0;
    g0[0] = 1u;                                           // count=1 (valid user D#)
    g0[1] = lds_byte;                                     // lds_addr
    g0[2] = (unsigned int)ga;                             // global_addr[31:0]
    g0[3] = (unsigned int)((ga >> 32) & 0x1FFFFFFull)     // global_addr[56:32]
          | (2u << 30);                                   // type=2 ("image")
    i32x8 g1;
    g1[0] = (int)((1u << 16)                              // data_size = 2 bytes
          | (1u << 20)                                    // pad_enable
          | (pad_interval_code << 22)
          | (pad_amount_code << 25));
    g1[1] = (int)((tile_d0 & 0xFFFFu) << 16);             // tensor_dim0[15:0]
    g1[2] = (int)(((tile_d0 >> 16) & 0xFFFFu)             // tensor_dim0[31:16]
          | ((tile_d1 & 0xFFFFu) << 16));                 // tensor_dim1[15:0]
    g1[3] = (int)(((tile_d1 >> 16) & 0xFFFFu)             // tensor_dim1[31:16]
          | (tile_d0 << 16));                             // tile_dim0
    g1[4] = (int)tile_d1;                                 // tile_dim1 (tile_dim2=0)
    g1[5] = (int)(unsigned int)stride0;                   // tensor_dim0_stride[31:0]
    g1[6] = (int)(unsigned int)((stride0 >> 32) & 0xFFFFull);
    g1[7] = 0;
    i32x4 z4 = {0, 0, 0, 0};
    i32x8 z8 = {0, 0, 0, 0, 0, 0, 0, 0};
    __builtin_amdgcn_tensor_load_to_lds(g0, g1, z4, z4, z8, 0);
}

// ---------------------------------------------------------------------------
// f32 -> f16 conversion
// ---------------------------------------------------------------------------
__global__ void cvt_f32_to_f16(const float* __restrict__ src,
                               _Float16* __restrict__ dst, int n) {
    int i = blockIdx.x * blockDim.x + threadIdx.x;
    int stride = gridDim.x * blockDim.x;
    for (; i < n; i += stride) dst[i] = (_Float16)src[i];
}

// ---------------------------------------------------------------------------
// GEMM core: block tile 256x128, 8 waves of 64x64, TDM double-buffered
// 32-wide K slabs.  LDS rows padded to 40 halves by TDM pad (codes 3/3).
// ---------------------------------------------------------------------------
#define GEMM_CORE(APTR, BPTR)                                                   \
    __shared__ __align__(16) _Float16 As[2][256][40];                           \
    __shared__ __align__(16) _Float16 Bs[2][128][40];                           \
    const int tid = threadIdx.x;                                                \
    const int lane = tid & 31, wave = tid >> 5;                                 \
    const int wm = wave & 3, wn = wave >> 2;                                    \
    const int half = lane >> 4, l16 = lane & 15;                                \
    const int m0 = blockIdx.x * 256;                                            \
    const int n0 = blockIdx.y * 128;                                            \
    v8f acc[4][4];                                                              \
    for (int i = 0; i < 4; ++i)                                                 \
        for (int j = 0; j < 4; ++j)                                             \
            for (int r = 0; r < 8; ++r) acc[i][j][r] = 0.0f;                    \
    if (wave == 0) {                                                            \
        tdm_load_2d(lds_lo32(&As[0][0][0]),                                     \
                    &APTR[(unsigned long long)m0 * Dm], 32, 256, Dm, 3, 3);     \
        tdm_load_2d(lds_lo32(&Bs[0][0][0]),                                     \
                    &BPTR[(unsigned long long)n0 * Dm], 32, 128, Dm, 3, 3);     \
    }                                                                           \
    for (int s = 0; s < Dm / 32; ++s) {                                         \
        const int cur = s & 1;                                                  \
        if (wave == 0) {                                                        \
            if (s + 1 < Dm / 32) {                                              \
                tdm_load_2d(lds_lo32(&As[cur ^ 1][0][0]),                       \
                            &APTR[(unsigned long long)m0 * Dm + (s + 1) * 32],  \
                            32, 256, Dm, 3, 3);                                 \
                tdm_load_2d(lds_lo32(&Bs[cur ^ 1][0][0]),                       \
                            &BPTR[(unsigned long long)n0 * Dm + (s + 1) * 32],  \
                            32, 128, Dm, 3, 3);                                 \
                __builtin_amdgcn_s_wait_tensorcnt(2);                           \
            } else {                                                            \
                __builtin_amdgcn_s_wait_tensorcnt(0);                           \
            }                                                                   \
        }                                                                       \
        __syncthreads();                                                        \
        F16x16 afr[4], bfr[4];                                                  \
        for (int i = 0; i < 4; ++i) {                                           \
            int r = wm * 64 + i * 16 + l16;                                     \
            afr[i].h[0] = *(const v8h*)&As[cur][r][half * 8];                   \
            afr[i].h[1] = *(const v8h*)&As[cur][r][half * 8 + 16];              \
        }                                                                       \
        for (int j = 0; j < 4; ++j) {                                           \
            int r = wn * 64 + j * 16 + l16;                                     \
            bfr[j].h[0] = *(const v8h*)&Bs[cur][r][half * 8];                   \
            bfr[j].h[1] = *(const v8h*)&Bs[cur][r][half * 8 + 16];              \
        }                                                                       \
        for (int i = 0; i < 4; ++i)                                             \
            for (int j = 0; j < 4; ++j)                                         \
                acc[i][j] = WMMA_F16(afr[i].v, bfr[j].v, acc[i][j]);            \
        __syncthreads();                                                        \
    }

// ---------------------------------------------------------------------------
// Fused QKV projection
// ---------------------------------------------------------------------------
__global__ __launch_bounds__(256, 1)
void qkv_gemm(const _Float16* __restrict__ X, const _Float16* __restrict__ W,
              const float* __restrict__ bq, const float* __restrict__ bk,
              const float* __restrict__ bv,
              _Float16* __restrict__ Qo, _Float16* __restrict__ Ko,
              _Float16* __restrict__ Vo) {
    GEMM_CORE(X, W)
    for (int i = 0; i < 4; ++i) {
        int mbase = m0 + wm * 64 + i * 16 + half * 8;
        for (int j = 0; j < 4; ++j) {
            int n = n0 + wn * 64 + j * 16 + l16;
            int which = n >> 10;
            int c = n & 1023;
            int h = c >> 6, d = c & 63;
            float bias = (which == 0) ? bq[c] : (which == 1) ? bk[c] : bv[c];
            float scale = (which == 0) ? 0.125f : 1.0f;   // fold 1/sqrt(Dh) into q
            _Float16* dst = (which == 0) ? Qo : (which == 1) ? Ko : Vo;
            for (int r = 0; r < 8; ++r) {
                int m = mbase + r;
                int bb = m >> 11, sq = m & 2047;
                unsigned long long off =
                    (((unsigned long long)(bb * Hn + h)) * Seq + sq) * Dh + d;
                dst[off] = (_Float16)((acc[i][j][r] + bias) * scale);
            }
        }
    }
}

// ---------------------------------------------------------------------------
// Output projection + bias + residual (f32 out)
// ---------------------------------------------------------------------------
__global__ __launch_bounds__(256, 1)
void out_proj(const _Float16* __restrict__ A_, const _Float16* __restrict__ W,
              const float* __restrict__ bo, const float* __restrict__ X,
              float* __restrict__ out) {
    GEMM_CORE(A_, W)
    for (int i = 0; i < 4; ++i) {
        int mbase = m0 + wm * 64 + i * 16 + half * 8;
        for (int j = 0; j < 4; ++j) {
            int n = n0 + wn * 64 + j * 16 + l16;
            float bias = bo[n];
            for (int r = 0; r < 8; ++r) {
                unsigned long long idx = (unsigned long long)(mbase + r) * Dm + n;
                out[idx] = acc[i][j][r] + bias + X[idx];
            }
        }
    }
}

// ---------------------------------------------------------------------------
// Flash attention per (b,h): 128-query tile, 64-key tiles streamed.
// Q/K tiles staged by TDM (pad 128B rows to 72 halves: codes 4/3);
// V transposed manually (TDM has no transpose).
// ---------------------------------------------------------------------------
__global__ __launch_bounds__(256, 1)
void attn_kernel(const _Float16* __restrict__ Qh, const _Float16* __restrict__ Kh,
                 const _Float16* __restrict__ Vh, _Float16* __restrict__ Oh) {
    __shared__ __align__(16) _Float16 Qs[128][72];
    __shared__ __align__(16) _Float16 Ks[64][72];
    __shared__ __align__(16) _Float16 Vt[64][72];      // [d][key]
    __shared__ __align__(16) _Float16 Ps[8][16][72];   // per-wave P tile

    const int tid = threadIdx.x, lane = tid & 31, wave = tid >> 5;
    const int half = lane >> 4, l16 = lane & 15;
    const int bh = blockIdx.y;
    const int q0 = blockIdx.x * 128;
    const unsigned long long base = (unsigned long long)bh * Seq * Dh;

    if (wave == 0)   // stage 128x64 Q tile via TDM
        tdm_load_2d(lds_lo32(&Qs[0][0]), &Qh[base + (unsigned long long)q0 * Dh],
                    64, 128, Dh, 4, 3);

    float m_i[8], l_i[8];
    for (int r = 0; r < 8; ++r) { m_i[r] = -1e30f; l_i[r] = 0.0f; }
    v8f acc_o[4];
    for (int j = 0; j < 4; ++j)
        for (int r = 0; r < 8; ++r) acc_o[j][r] = 0.0f;

    for (int kt = 0; kt < Seq; kt += 64) {
        __syncthreads();
        if (wave == 0)   // K tile via TDM
            tdm_load_2d(lds_lo32(&Ks[0][0]), &Kh[base + (unsigned long long)kt * Dh],
                        64, 64, Dh, 4, 3);
        {   // V tile transposed, all threads
            int r = tid >> 2;
            int c = (tid & 3) * 16;
            const _Float16* vsrc = &Vh[base + (unsigned long long)(kt + r) * Dh + c];
            for (int j = 0; j < 16; ++j) Vt[c + j][r] = vsrc[j];
        }
        if (wave == 0) __builtin_amdgcn_s_wait_tensorcnt(0);
        __syncthreads();

        // S = Q * K^T
        v8f sc[4];
        for (int j = 0; j < 4; ++j)
            for (int r = 0; r < 8; ++r) sc[j][r] = 0.0f;
        for (int ks = 0; ks < 2; ++ks) {
            F16x16 afr;
            int r = wave * 16 + l16;
            afr.h[0] = *(const v8h*)&Qs[r][ks * 32 + half * 8];
            afr.h[1] = *(const v8h*)&Qs[r][ks * 32 + half * 8 + 16];
            for (int j = 0; j < 4; ++j) {
                F16x16 bfr;
                int n = j * 16 + l16;
                bfr.h[0] = *(const v8h*)&Ks[n][ks * 32 + half * 8];
                bfr.h[1] = *(const v8h*)&Ks[n][ks * 32 + half * 8 + 16];
                sc[j] = WMMA_F16(afr.v, bfr.v, sc[j]);
            }
        }

        // streaming softmax
        for (int r = 0; r < 8; ++r) {
            float rmax = fmaxf(fmaxf(sc[0][r], sc[1][r]), fmaxf(sc[2][r], sc[3][r]));
            for (int msk = 1; msk < 16; msk <<= 1)
                rmax = fmaxf(rmax, __shfl_xor(rmax, msk, 32));
            float newm = fmaxf(m_i[r], rmax);
            float alpha = __expf(m_i[r] - newm);
            m_i[r] = newm;
            float rsum = 0.0f;
            for (int j = 0; j < 4; ++j) {
                float p = __expf(sc[j][r] - newm);
                rsum += p;
                Ps[wave][r + 8 * half][j * 16 + l16] = (_Float16)p;
            }
            for (int msk = 1; msk < 16; msk <<= 1)
                rsum += __shfl_xor(rsum, msk, 32);
            l_i[r] = l_i[r] * alpha + rsum;
            for (int j = 0; j < 4; ++j) acc_o[j][r] *= alpha;
        }
        asm volatile("s_wait_dscnt 0" ::: "memory");

        // O += P * V
        for (int ks = 0; ks < 2; ++ks) {
            F16x16 afr;
            afr.h[0] = *(const v8h*)&Ps[wave][l16][ks * 32 + half * 8];
            afr.h[1] = *(const v8h*)&Ps[wave][l16][ks * 32 + half * 8 + 16];
            for (int j = 0; j < 4; ++j) {
                F16x16 bfr;
                int d = j * 16 + l16;
                bfr.h[0] = *(const v8h*)&Vt[d][ks * 32 + half * 8];
                bfr.h[1] = *(const v8h*)&Vt[d][ks * 32 + half * 8 + 16];
                acc_o[j] = WMMA_F16(afr.v, bfr.v, acc_o[j]);
            }
        }
    }

    const int b = bh >> 4, h = bh & 15;
    for (int r = 0; r < 8; ++r) {
        float inv = 1.0f / l_i[r];
        int srow = q0 + wave * 16 + r + 8 * half;
        for (int j = 0; j < 4; ++j) {
            int d = j * 16 + l16;
            Oh[(unsigned long long)(b * Seq + srow) * Dm + h * Dh + d] =
                (_Float16)(acc_o[j][r] * inv);
        }
    }
}

// ---------------------------------------------------------------------------
// Host-side launch
// ---------------------------------------------------------------------------
extern "C" void kernel_launch(void* const* d_in, const int* in_sizes, int n_in,
                              void* d_out, int out_size, void* d_ws, size_t ws_size,
                              hipStream_t stream) {
    const float* x  = (const float*)d_in[0];
    const float* Wq = (const float*)d_in[1];
    const float* bq = (const float*)d_in[2];
    const float* Wk = (const float*)d_in[3];
    const float* bk = (const float*)d_in[4];
    const float* Wv = (const float*)d_in[5];
    const float* bv = (const float*)d_in[6];
    const float* Wo = (const float*)d_in[7];
    const float* bo = (const float*)d_in[8];

    _Float16* ws   = (_Float16*)d_ws;
    _Float16* xh   = ws + XH_OFF;
    _Float16* wqkv = ws + WQKV_OFF;
    _Float16* woh  = ws + WO_OFF;
    _Float16* qh   = ws + QH_OFF;
    _Float16* kh   = ws + KH_OFF;
    _Float16* vh   = ws + VH_OFF;
    _Float16* ah   = ws + AH_OFF;

    const int NW = Dm * Dm;
    const int NX = Mrows * Dm;
    cvt_f32_to_f16<<<2048, 256, 0, stream>>>(x,  xh, NX);
    cvt_f32_to_f16<<<1024, 256, 0, stream>>>(Wq, wqkv,                 NW);
    cvt_f32_to_f16<<<1024, 256, 0, stream>>>(Wk, wqkv + (size_t)NW,    NW);
    cvt_f32_to_f16<<<1024, 256, 0, stream>>>(Wv, wqkv + 2*(size_t)NW,  NW);
    cvt_f32_to_f16<<<1024, 256, 0, stream>>>(Wo, woh, NW);

    qkv_gemm<<<dim3(Mrows / 256, 3072 / 128), 256, 0, stream>>>(
        xh, wqkv, bq, bk, bv, qh, kh, vh);

    attn_kernel<<<dim3(Seq / 128, Bsz * Hn), 256, 0, stream>>>(qh, kh, vh, ah);

    out_proj<<<dim3(Mrows / 256, Dm / 128), 256, 0, stream>>>(
        ah, woh, bo, x, (float*)d_out);
}
